// BiLSTM_CRF_DAE_44281112821767
// MI455X (gfx1250) — matrix operand order repaired and checked by reading, not compiled, not run on hardware
//
#include <hip/hip_runtime.h>
#include <stdint.h>

// ---------------- constants (from reference) ----------------
#define TT   256          // sequence length
#define BB   64           // batch
#define EE   256          // embed dim
#define HH   256          // hidden
#define GG   1024         // 4*H gates
#define HID  512          // 2*H
#define NTAG 34
#define NTAGP 48          // NTAG padded to 3 WMMA tiles
#define TB   (TT*BB)      // 16384 rows

// ---------------- types for WMMA ----------------
typedef __bf16 bf16_t;
typedef __attribute__((ext_vector_type(16))) __bf16 bf16x16;
typedef __attribute__((ext_vector_type(8)))  float  f32x8;

union FragU {
    uint4   q[2];   // 32 bytes = 16 bf16
    bf16x16 v;
};

__device__ __forceinline__ f32x8 wmma_bb(const FragU& a, const FragU& b, f32x8 c) {
    // D(16x16 f32) = A(16x32 bf16) * B(32x16 bf16) + C
    return __builtin_amdgcn_wmma_f32_16x16x32_bf16(
        /*neg_a=*/false, a.v, /*neg_b=*/false, b.v,
        /*c_mod=*/(short)0, c, /*reuse_a=*/false, /*reuse_b=*/false);
}

__device__ __forceinline__ unsigned short f32_to_bf16(float f) {
    unsigned int u = __float_as_uint(f);
    unsigned int r = 0x7FFFu + ((u >> 16) & 1u);     // round-to-nearest-even
    return (unsigned short)((u + r) >> 16);
}

__device__ __forceinline__ float sigmf(float z) { return 1.0f / (1.0f + __expf(-z)); }

// A fragment: storage is row-major (row, K), lane l: M = mrow, two b128 loads.
// 16-bit A layout: vgpr v -> K = (v>=4?16:0) + (lane/16)*8 + (v%4)*2
__device__ __forceinline__ void load_afrag(FragU& f, const unsigned short* base,
                                           int K, int row, int kb, int khalf) {
    const unsigned short* p = base + (size_t)row * K + kb + khalf * 8;
    f.q[0] = *reinterpret_cast<const uint4*>(p);
    f.q[1] = *reinterpret_cast<const uint4*>(p + 16);
}
// B fragment: W stored row-major (N, K), lane indexes N column of B = row of W.
// 16-bit B layout: vgpr v -> K = (v>=4?16:0) + (lane/16)*8 + (v%4)*2
__device__ __forceinline__ void load_bfrag(FragU& f, const unsigned short* base,
                                           int K, int nrow, int kb, int khalf) {
    const unsigned short* p = base + (size_t)nrow * K + kb + khalf * 8;
    f.q[0] = *reinterpret_cast<const uint4*>(p);
    f.q[1] = *reinterpret_cast<const uint4*>(p + 16);
}

// ---------------- small prep kernels ----------------
__global__ void cvt_f32_bf16_kernel(const float* __restrict__ src,
                                    unsigned short* __restrict__ dst,
                                    int srcN, int totalN) {
    int i = blockIdx.x * blockDim.x + threadIdx.x;
    if (i < totalN) dst[i] = (i < srcN) ? f32_to_bf16(src[i]) : (unsigned short)0;
}

__global__ void bias_comb_kernel(const float* __restrict__ a, const float* __restrict__ b,
                                 float* __restrict__ dst, int n) {
    int i = blockIdx.x * blockDim.x + threadIdx.x;
    if (i < n) dst[i] = a[i] + b[i];
}

__global__ void embed_bf16_kernel(const int* __restrict__ x, const float* __restrict__ emb,
                                  unsigned short* __restrict__ A0) {
    int i = blockIdx.x * blockDim.x + threadIdx.x;   // over TB*EE
    if (i >= TB * EE) return;
    int m = i >> 8, e = i & 255;
    int tok = x[m];
    A0[i] = f32_to_bf16(emb[(size_t)tok * EE + e]);
}

// ---------------- xp GEMM: out[dir][m][g] = A[m,:] . W[dir][g,:] + bias[dir][g] ----
// block = 256 thr (8 waves); wave tile = 32(M) x 64(N); block tile = 128x128
__global__ __launch_bounds__(256) void gemm_xp_kernel(
        const unsigned short* __restrict__ A,    // TB x K bf16
        const unsigned short* __restrict__ W,    // 2 x GG x K bf16
        const float* __restrict__ bias,          // 2 x GG (b_ih+b_hh)
        float* __restrict__ out,                 // 2 x TB x GG
        int K) {
    int dir = blockIdx.z;
    const unsigned short* Wd = W + (size_t)dir * GG * K;
    const float* bd = bias + dir * GG;
    float* od = out + (size_t)dir * TB * GG;

    int wave = threadIdx.x >> 5, lane = threadIdx.x & 31;
    int khalf = lane >> 4, l15 = lane & 15;
    int mwave = wave >> 1, nwave = wave & 1;
    int mbase = blockIdx.x * 128 + mwave * 32;
    int nbase = blockIdx.y * 128 + nwave * 64;

    f32x8 zero8 = {0.f,0.f,0.f,0.f,0.f,0.f,0.f,0.f};
    f32x8 acc[2][4];
    for (int i = 0; i < 2; ++i) for (int j = 0; j < 4; ++j) acc[i][j] = zero8;

    for (int kb = 0; kb < K; kb += 32) {
        FragU a[2];
        for (int mt = 0; mt < 2; ++mt)
            load_afrag(a[mt], A, K, mbase + mt * 16 + l15, kb, khalf);
        for (int nt = 0; nt < 4; ++nt) {
            FragU bf;
            load_bfrag(bf, Wd, K, nbase + nt * 16 + l15, kb, khalf);
            for (int mt = 0; mt < 2; ++mt)
                acc[mt][nt] = wmma_bb(a[mt], bf, acc[mt][nt]);
        }
    }
    for (int mt = 0; mt < 2; ++mt)
        for (int nt = 0; nt < 4; ++nt)
            for (int r = 0; r < 8; ++r) {
                int row = mbase + mt * 16 + khalf * 8 + r;   // C: M = r + 8*(lane/16)
                int col = nbase + nt * 16 + l15;             // C: N = lane%16
                od[(size_t)row * GG + col] = acc[mt][nt][r] + bd[col];
            }
}

// ---------------- persistent LSTM scan (one block per direction) ----------------
// 1024 thr = 32 waves. wave -> (mhalf, chunk): 2 M-tiles x 4 gate-tiles at h cols
// chunk*16..+15. h kept in LDS as bf16 (A matrix); c kept in registers per lane.
__global__ __launch_bounds__(1024) void lstm_scan_kernel(
        const float* __restrict__ xp,            // 2 x TB x GG (precomputed, + biases)
        const unsigned short* __restrict__ Whh,  // 2 x GG x HH bf16
        const int* __restrict__ x,               // TT x BB tokens (mask = !=0)
        unsigned short* __restrict__ Aout,       // TB x outStride bf16 (next layer input)
        int outStride) {
    int dir = blockIdx.x;
    const float* xpd = xp + (size_t)dir * TB * GG;
    const unsigned short* Wd = Whh + (size_t)dir * GG * HH;
    int colOff = dir * HH;                       // fwd -> cols [0,256), bwd -> [256,512)

    __shared__ unsigned short hbuf[BB * HH];     // 32 KB carried hidden state (bf16)
    for (int i = threadIdx.x; i < BB * HH; i += 1024) hbuf[i] = 0;
    __syncthreads();

    int wave = threadIdx.x >> 5, lane = threadIdx.x & 31;
    int chunk = wave & 15;                       // h column chunk (16 cols per gate)
    int mhalf = wave >> 4;                       // batch half (rows 32*mhalf..+31)
    int khalf = lane >> 4, l15 = lane & 15;
    int hidx = chunk * 16 + l15;

    float cst[2][8];
    for (int i = 0; i < 2; ++i) for (int r = 0; r < 8; ++r) cst[i][r] = 0.f;
    f32x8 zero8 = {0.f,0.f,0.f,0.f,0.f,0.f,0.f,0.f};

    for (int step = 0; step < TT; ++step) {
        int t = dir ? (TT - 1 - step) : step;

        f32x8 acc[2][4];
        for (int i = 0; i < 2; ++i) for (int g = 0; g < 4; ++g) acc[i][g] = zero8;

        // g += h @ Whh^T : A = hbuf (64 x 256 bf16 in LDS), B = Whh rows
        for (int kb = 0; kb < HH; kb += 32) {
            FragU a[2];
            for (int mt = 0; mt < 2; ++mt) {
                int row = mhalf * 32 + mt * 16 + l15;
                const unsigned short* p = &hbuf[row * HH + kb + khalf * 8];
                a[mt].q[0] = *reinterpret_cast<const uint4*>(p);
                a[mt].q[1] = *reinterpret_cast<const uint4*>(p + 16);
            }
            for (int g = 0; g < 4; ++g) {
                FragU bf;
                load_bfrag(bf, Wd, HH, g * HH + chunk * 16 + l15, kb, khalf);
                for (int mt = 0; mt < 2; ++mt)
                    acc[mt][g] = wmma_bb(a[mt], bf, acc[mt][g]);
            }
        }
        __syncthreads();   // all reads of hbuf done before masked update

        if (step + 1 < TT) {  // prefetch next timestep's xp slab into L2
            int tn = dir ? (t - 1) : (t + 1);
            __builtin_prefetch(xpd + ((size_t)tn * BB) * GG, 0, 1);
        }

        for (int mt = 0; mt < 2; ++mt)
            for (int r = 0; r < 8; ++r) {
                int b = mhalf * 32 + mt * 16 + khalf * 8 + r;  // C-layout row
                const float* xr = xpd + ((size_t)t * BB + b) * GG;
                float gi = acc[mt][0][r] + xr[0   + hidx];
                float gf = acc[mt][1][r] + xr[256 + hidx];
                float gg = acc[mt][2][r] + xr[512 + hidx];
                float go = acc[mt][3][r] + xr[768 + hidx];
                float cn = sigmf(gf) * cst[mt][r] + sigmf(gi) * tanhf(gg);
                float hn = sigmf(go) * tanhf(cn);
                bool  m  = (x[t * BB + b] != 0);
                float hout = 0.f;
                if (m) {
                    cst[mt][r] = cn;
                    hbuf[b * HH + hidx] = f32_to_bf16(hn);  // carry; else keep old
                    hout = hn;
                }
                Aout[((size_t)t * BB + b) * outStride + colOff + hidx] = f32_to_bf16(hout);
            }
        __syncthreads();   // next step's A reads see updated hbuf
    }
}

// ---------------- emissions GEMM: em[m][n] = A2[m,:] . Wlin[n,:] + b_lin[n] -------
__global__ __launch_bounds__(256) void emis_kernel(
        const unsigned short* __restrict__ A2,    // TB x HID bf16
        const unsigned short* __restrict__ Wlin,  // NTAGP x HID bf16 (rows >=34 zero)
        const float* __restrict__ b_lin,
        float* __restrict__ em) {                 // TB x NTAG
    int wave = threadIdx.x >> 5, lane = threadIdx.x & 31;
    int khalf = lane >> 4, l15 = lane & 15;
    int mbase = blockIdx.x * 128 + wave * 16;

    f32x8 zero8 = {0.f,0.f,0.f,0.f,0.f,0.f,0.f,0.f};
    f32x8 acc[3] = {zero8, zero8, zero8};

    for (int kb = 0; kb < HID; kb += 32) {
        FragU a;
        load_afrag(a, A2, HID, mbase + l15, kb, khalf);
        for (int nt = 0; nt < 3; ++nt) {
            FragU bf;
            load_bfrag(bf, Wlin, HID, nt * 16 + l15, kb, khalf);
            acc[nt] = wmma_bb(a, bf, acc[nt]);
        }
    }
    for (int nt = 0; nt < 3; ++nt)
        for (int r = 0; r < 8; ++r) {
            int row = mbase + khalf * 8 + r;
            int col = nt * 16 + l15;
            if (col < NTAG)
                em[(size_t)row * NTAG + col] = acc[nt][r] + b_lin[col];
        }
}

// ---------------- Viterbi decode: one block per batch element ----------------
__global__ __launch_bounds__(64) void viterbi_kernel(
        const float* __restrict__ em, const int* __restrict__ x,
        const float* __restrict__ startt, const float* __restrict__ endt,
        const float* __restrict__ trans,
        int* __restrict__ hist,                  // TT x BB x NTAG backpointers
        int* __restrict__ out) {                 // TT x BB tags
    int b = blockIdx.x, n = threadIdx.x;
    __shared__ float sc[NTAG];
    __shared__ float tr[NTAG * NTAG];
    for (int i = n; i < NTAG * NTAG; i += 64) tr[i] = trans[i];
    if (n < NTAG) sc[n] = startt[n] + em[(size_t)b * NTAG + n];
    __syncthreads();

    for (int t = 1; t < TT; ++t) {
        float ns = 0.f; int na = 0; bool act = (n < NTAG);
        if (act) {
            float best = -3.0e38f; int arg = 0;
            for (int p = 0; p < NTAG; ++p) {
                float s = sc[p] + tr[p * NTAG + n];
                if (s > best) { best = s; arg = p; }   // first max, like argmax
            }
            bool m = (x[t * BB + b] != 0);
            ns = m ? best + em[((size_t)t * BB + b) * NTAG + n] : sc[n];
            na = m ? arg : n;
            hist[((size_t)t * BB + b) * NTAG + n] = na;
        }
        __syncthreads();
        if (act) sc[n] = ns;
        __syncthreads();
    }

    if (n == 0) {
        float best = -3.0e38f; int tag = 0;
        for (int k = 0; k < NTAG; ++k) {
            float s = sc[k] + endt[k];
            if (s > best) { best = s; tag = k; }
        }
        out[(TT - 1) * BB + b] = (x[(TT - 1) * BB + b] != 0) ? tag : 0;
        for (int t = TT - 2; t >= 0; --t) {
            tag = hist[((size_t)(t + 1) * BB + b) * NTAG + tag];
            out[t * BB + b] = (x[t * BB + b] != 0) ? tag : 0;
        }
    }
}

// ---------------- launch ----------------
extern "C" void kernel_launch(void* const* d_in, const int* in_sizes, int n_in,
                              void* d_out, int out_size, void* d_ws, size_t ws_size,
                              hipStream_t stream) {
    (void)in_sizes; (void)n_in; (void)out_size; (void)ws_size;
    const int*   x        = (const int*)  d_in[0];
    const float* embed    = (const float*)d_in[2];
    const float* w_ih_l0  = (const float*)d_in[3];
    const float* w_hh_l0  = (const float*)d_in[4];
    const float* b_ih_l0  = (const float*)d_in[5];
    const float* b_hh_l0  = (const float*)d_in[6];
    const float* w_ih_l1  = (const float*)d_in[7];
    const float* w_hh_l1  = (const float*)d_in[8];
    const float* b_ih_l1  = (const float*)d_in[9];
    const float* b_hh_l1  = (const float*)d_in[10];
    const float* w_lin    = (const float*)d_in[11];
    const float* b_lin    = (const float*)d_in[12];
    const float* start_tr = (const float*)d_in[13];
    const float* end_tr   = (const float*)d_in[14];
    const float* trans    = (const float*)d_in[15];

    char* w = (char*)d_ws;
    auto alloc = [&](size_t bytes) { char* p = w; w += (bytes + 255) & ~(size_t)255; return p; };
    unsigned short* A0    = (unsigned short*)alloc((size_t)TB * EE  * 2);
    unsigned short* A1    = (unsigned short*)alloc((size_t)TB * HID * 2);
    unsigned short* A2    = (unsigned short*)alloc((size_t)TB * HID * 2);
    float*          XP    = (float*)         alloc((size_t)2 * TB * GG * 4);
    unsigned short* WIH0  = (unsigned short*)alloc((size_t)2 * GG * EE  * 2);
    unsigned short* WIH1  = (unsigned short*)alloc((size_t)2 * GG * HID * 2);
    unsigned short* WHH0  = (unsigned short*)alloc((size_t)2 * GG * HH * 2);
    unsigned short* WHH1  = (unsigned short*)alloc((size_t)2 * GG * HH * 2);
    unsigned short* WLIN  = (unsigned short*)alloc((size_t)NTAGP * HID * 2);
    float*          BIAS0 = (float*)         alloc((size_t)2 * GG * 4);
    float*          BIAS1 = (float*)         alloc((size_t)2 * GG * 4);
    float*          EM    = (float*)         alloc((size_t)TB * NTAG * 4);
    int*            HIST  = (int*)           alloc((size_t)TT * BB * NTAG * 4);

    auto blks = [](int n) { return (n + 255) / 256; };

    // weight conversion fp32 -> bf16 (WMMA operand precision)
    cvt_f32_bf16_kernel<<<blks(2*GG*EE),  256, 0, stream>>>(w_ih_l0, WIH0, 2*GG*EE,  2*GG*EE);
    cvt_f32_bf16_kernel<<<blks(2*GG*HID), 256, 0, stream>>>(w_ih_l1, WIH1, 2*GG*HID, 2*GG*HID);
    cvt_f32_bf16_kernel<<<blks(2*GG*HH),  256, 0, stream>>>(w_hh_l0, WHH0, 2*GG*HH,  2*GG*HH);
    cvt_f32_bf16_kernel<<<blks(2*GG*HH),  256, 0, stream>>>(w_hh_l1, WHH1, 2*GG*HH,  2*GG*HH);
    cvt_f32_bf16_kernel<<<blks(NTAGP*HID),256, 0, stream>>>(w_lin,  WLIN, NTAG*HID, NTAGP*HID);
    bias_comb_kernel<<<blks(2*GG), 256, 0, stream>>>(b_ih_l0, b_hh_l0, BIAS0, 2*GG);
    bias_comb_kernel<<<blks(2*GG), 256, 0, stream>>>(b_ih_l1, b_hh_l1, BIAS1, 2*GG);

    // embedding gather -> bf16 layer-0 input
    embed_bf16_kernel<<<blks(TB*EE), 256, 0, stream>>>(x, embed, A0);

    dim3 gx(TB / 128, GG / 128, 2);   // (128, 8, 2)
    // layer 0
    gemm_xp_kernel<<<gx, 256, 0, stream>>>(A0, WIH0, BIAS0, XP, EE);
    lstm_scan_kernel<<<2, 1024, 0, stream>>>(XP, WHH0, x, A1, HID);
    // layer 1
    gemm_xp_kernel<<<gx, 256, 0, stream>>>(A1, WIH1, BIAS1, XP, HID);
    lstm_scan_kernel<<<2, 1024, 0, stream>>>(XP, WHH1, x, A2, HID);
    // emissions + CRF Viterbi
    emis_kernel<<<TB / 128, 256, 0, stream>>>(A2, WLIN, b_lin, EM);
    viterbi_kernel<<<BB, 64, 0, stream>>>(EM, x, start_tr, end_tr, trans, HIST, (int*)d_out);
}
